// SAMImageEncoder_40450001994073
// MI455X (gfx1250) — matrix operand order repaired and checked by reading, not compiled, not run on hardware
//
#include <hip/hip_runtime.h>
#include <hip/hip_bf16.h>
#include <math.h>

typedef __bf16 bf16_t;
typedef __attribute__((ext_vector_type(16))) __bf16 v16bf;
typedef __attribute__((ext_vector_type(8)))  float  v8f;
typedef __attribute__((ext_vector_type(4)))  int    v4i;

__device__ __forceinline__ bf16_t f2bf(float f) { return (bf16_t)f; }
__device__ __forceinline__ float  bf2f(bf16_t b) { return (float)b; }

// CDNA5 async global->LDS path (ASYNCcnt + s_wait_asynccnt), guarded so the
// build falls back to plain b128 copies if the toolchain lacks the builtins.
// Probe: param 0 is 'int4 __device__ *' => AS1 v4i*; dst symmetric => AS3 v4i*.
#if __has_builtin(__builtin_amdgcn_global_load_async_to_lds_b128) && \
    __has_builtin(__builtin_amdgcn_s_wait_asynccnt)
#define A_ASYNC 1
typedef __attribute__((address_space(1))) v4i as1_v4i;
typedef __attribute__((address_space(3))) v4i as3_v4i;
#else
#define A_ASYNC 0
#endif

// ---------------------------------------------------------------------------
// Generic bf16 WMMA GEMM:  C = epilogue( A(bf16, MxK) @ W(KxN) )
//   BM=128, BN=64, BK=32; 256 threads = 8 waves; each wave owns a 32x32 tile
//   => 2x2 v_wmma_f32_16x16x32_bf16 per K-step.
// Batched via blockIdx.z = b*nH + h with separate b/h strides (attention).
// ---------------------------------------------------------------------------
constexpr int BM = 128, BN = 64, BK = 32;
constexpr int LDS_STRIDE = BK + 8; // 40 elems (80B) -> 16B-aligned rows, pad vs bank conflicts

template <typename WT, bool TRANSW, bool OUTBF, bool DOGELU>
__global__ __launch_bounds__(256) void k_gemm(
    const bf16_t* __restrict__ A, long lda, long sAb, long sAh,
    const WT* __restrict__ W, long ldw, long sWb, long sWh,
    const float* __restrict__ bias, const float* __restrict__ pos,
    const float* __restrict__ resid,
    void* __restrict__ Cv, long ldc, long sCb, long sCh,
    int M, int Nt, int K, int nH)
{
  __shared__ bf16_t As[BM][LDS_STRIDE];
  __shared__ bf16_t Bs[BN][LDS_STRIDE];

  const int tid  = threadIdx.x;
  const int lane = tid & 31;
  const int wv   = tid >> 5;
  const int wm   = wv & 3;   // wave position along M (4)
  const int wn   = wv >> 2;  // wave position along N (2)
  const int m0   = blockIdx.y * BM;
  const int n0   = blockIdx.x * BN;
  const int zb   = blockIdx.z / nH;
  const int zh   = blockIdx.z % nH;

  const bf16_t* Ab = A + (long)zb * sAb + (long)zh * sAh;
  const WT*     Wb = W + (long)zb * sWb + (long)zh * sWh;

  v8f acc[2][2] = {};

  const int arow = tid >> 1;        // 0..127
  const int akb  = (tid & 1) * 16;  // 0 or 16

  const int khalf = lane >> 4;      // 0|1
  const int l16   = lane & 15;

  const int nK = K / BK;
  for (int kk = 0; kk < nK; ++kk) {
    const int k0 = kk * BK;

    // ---- stage A tile (128x32 bf16), 16 elems (= 2x b128) per thread ----
    {
      const bf16_t* srcp = Ab + (long)(m0 + arow) * lda + k0 + akb;
#if A_ASYNC
      as1_v4i* g1 = (as1_v4i*)const_cast<bf16_t*>(srcp);
      as3_v4i* l3 = (as3_v4i*)(&As[arow][akb]);
      // INST_OFFSET applies to both global and LDS address -> one base, two offsets
      __builtin_amdgcn_global_load_async_to_lds_b128(g1, l3, 0, 0);
      __builtin_amdgcn_global_load_async_to_lds_b128(g1, l3, 16, 0);
#else
      const uint4* src = (const uint4*)srcp;
      uint4 u0 = src[0];
      uint4 u1 = src[1];
      *(uint4*)(&As[arow][akb])     = u0;
      *(uint4*)(&As[arow][akb + 8]) = u1;
#endif
      if (kk + 1 < nK)  // gfx1250 global_prefetch for next K tile
        __builtin_prefetch((const void*)(srcp + BK), 0, 1);
    }

    // ---- stage W tile into Bs[n][k] (N-major so B-frags are contiguous) ----
    // N-edge: Nt and colb are always multiples of 8 here, so clamping the
    // source address to Nt-8 only ever duplicates data into Bs rows whose
    // output columns are >= Nt (discarded by the epilogue) -> no branches.
    if constexpr (!TRANSW) {
      const int wk  = tid >> 3;        // 0..31 (k)
      const int wnb = (tid & 7) * 8;   // 0..56 (n)
      int colb = n0 + wnb;
      if (colb + 8 > Nt) colb = Nt - 8;
      if constexpr (sizeof(WT) == 4) {  // f32 weights, convert to bf16
        const float* wp = (const float*)Wb + (long)(k0 + wk) * ldw + colb;
        const float4 f0 = ((const float4*)wp)[0];
        const float4 f1 = ((const float4*)wp)[1];
        Bs[wnb + 0][wk] = f2bf(f0.x); Bs[wnb + 1][wk] = f2bf(f0.y);
        Bs[wnb + 2][wk] = f2bf(f0.z); Bs[wnb + 3][wk] = f2bf(f0.w);
        Bs[wnb + 4][wk] = f2bf(f1.x); Bs[wnb + 5][wk] = f2bf(f1.y);
        Bs[wnb + 6][wk] = f2bf(f1.z); Bs[wnb + 7][wk] = f2bf(f1.w);
      } else {                          // bf16 weights (attention V)
        union { uint4 u; bf16_t h[8]; } tv;
        tv.u = *(const uint4*)((const bf16_t*)Wb + (long)(k0 + wk) * ldw + colb);
        #pragma unroll
        for (int j = 0; j < 8; ++j) Bs[wnb + j][wk] = tv.h[j];
      }
    } else {  // W accessed transposed (Q*K^T): (k,n) = W[n*ldw+k]; N=1024 exact
      const int wn_ = tid >> 2;         // 0..63 (n)
      const int wkb = (tid & 3) * 8;    // 0,8,16,24 (k)
      const uint4 u = *(const uint4*)((const bf16_t*)Wb + (long)(n0 + wn_) * ldw + k0 + wkb);
      *(uint4*)(&Bs[wn_][wkb]) = u;
    }

#if A_ASYNC
    __builtin_amdgcn_s_wait_asynccnt(0);
#endif
    __syncthreads();

    // ---- load fragments per ISA 16-bit layouts, 4x WMMA ----
    union Frag { v16bf v; uint4 u[2]; };
    Frag af[2], bf[2];
    #pragma unroll
    for (int mi = 0; mi < 2; ++mi) {
      const int r = wm * 32 + mi * 16 + l16;
      // lanes 0-15: K 0..7 & 16..23 ; lanes 16-31: K 8..15 & 24..31
      af[mi].u[0] = *(const uint4*)(&As[r][khalf * 8]);
      af[mi].u[1] = *(const uint4*)(&As[r][16 + khalf * 8]);
    }
    #pragma unroll
    for (int ni = 0; ni < 2; ++ni) {
      const int c = wn * 32 + ni * 16 + l16;
      // lanes 0-15: K 0..7 & 16..23 ; lanes 16-31: K 8..15 & 24..31
      bf[ni].u[0] = *(const uint4*)(&Bs[c][khalf * 8]);
      bf[ni].u[1] = *(const uint4*)(&Bs[c][16 + khalf * 8]);
    }
    #pragma unroll
    for (int mi = 0; mi < 2; ++mi)
      #pragma unroll
      for (int ni = 0; ni < 2; ++ni)
        acc[mi][ni] = __builtin_amdgcn_wmma_f32_16x16x32_bf16(
            false, af[mi].v, false, bf[ni].v, (short)0, acc[mi][ni], false, false);

    __syncthreads();
  }

  // ---- epilogue: C/D layout => vgpr r holds row (khalf*8 + r), col = lane%16 ----
  float*  Cf = (float*)Cv;
  bf16_t* Ch = (bf16_t*)Cv;
  const long cbase = (long)zb * sCb + (long)zh * sCh;
  #pragma unroll
  for (int mi = 0; mi < 2; ++mi) {
    #pragma unroll
    for (int ni = 0; ni < 2; ++ni) {
      const int colg = n0 + wn * 32 + ni * 16 + l16;
      if (colg >= Nt) continue;
      #pragma unroll
      for (int r = 0; r < 8; ++r) {
        const int rowg = m0 + wm * 32 + mi * 16 + khalf * 8 + r;
        float v = acc[mi][ni][r];
        if (bias)  v += bias[colg];
        if (pos)   v += pos[(long)(rowg & 1023) * Nt + colg];
        if constexpr (DOGELU) v = 0.5f * v * (1.f + erff(v * 0.70710678118654752f));
        if (resid) v += resid[(long)rowg * ldc + colg];
        const long idx = cbase + (long)rowg * ldc + colg;
        if constexpr (OUTBF) Ch[idx] = f2bf(v);
        else                 Cf[idx] = v;
      }
    }
  }
}

// ---------------------------------------------------------------------------
// Patchify: x(B,3,512,512) -> bf16 rows (B*1024, 768) [c*256 + py*16 + px]
// ---------------------------------------------------------------------------
__global__ void k_im2row(const float* __restrict__ x, bf16_t* __restrict__ o, int total)
{
  int i = blockIdx.x * 256 + threadIdx.x;
  if (i >= total) return;
  int col = i % 768;
  int n   = (i / 768) & 1023;
  int b   = i / (768 * 1024);
  int c = col >> 8, rem = col & 255;
  int py = rem >> 4, px = rem & 15;
  int gy = n >> 5, gx = n & 31;
  o[i] = f2bf(x[(((long)(b * 3 + c) * 512) + gy * 16 + py) * 512 + gx * 16 + px]);
}

// ---------------------------------------------------------------------------
// LayerNorm over D=768 (one 256-thread block per row), bf16 output
// ---------------------------------------------------------------------------
__global__ __launch_bounds__(256) void k_ln(const float* __restrict__ x,
                                            const float* __restrict__ w,
                                            const float* __restrict__ b,
                                            bf16_t* __restrict__ o)
{
  __shared__ float red[256];
  const int row = blockIdx.x, t = threadIdx.x;
  const float* xr = x + (long)row * 768;
  float v0 = xr[t], v1 = xr[t + 256], v2 = xr[t + 512];
  red[t] = v0 + v1 + v2;
  __syncthreads();
  for (int off = 128; off > 0; off >>= 1) { if (t < off) red[t] += red[t + off]; __syncthreads(); }
  const float mean = red[0] * (1.f / 768.f);
  __syncthreads();
  const float d0 = v0 - mean, d1 = v1 - mean, d2 = v2 - mean;
  red[t] = d0 * d0 + d1 * d1 + d2 * d2;
  __syncthreads();
  for (int off = 128; off > 0; off >>= 1) { if (t < off) red[t] += red[t + off]; __syncthreads(); }
  const float rstd = rsqrtf(red[0] * (1.f / 768.f) + 1e-5f);
  bf16_t* orow = o + (long)row * 768;
  orow[t]       = f2bf(d0 * rstd * w[t]       + b[t]);
  orow[t + 256] = f2bf(d1 * rstd * w[t + 256] + b[t + 256]);
  orow[t + 512] = f2bf(d2 * rstd * w[t + 512] + b[t + 512]);
}

// ---------------------------------------------------------------------------
// Softmax over 1024 bf16 logits per row; one wave32 per row, scale fused.
// ---------------------------------------------------------------------------
__global__ __launch_bounds__(256) void k_softmax(bf16_t* __restrict__ S, int rows)
{
  const int wv = threadIdx.x >> 5, lane = threadIdx.x & 31;
  const int row = blockIdx.x * 8 + wv;
  if (row >= rows) return;
  bf16_t* p = S + (long)row * 1024;
  float v[32];
  float mx = -3.4e38f;
  #pragma unroll
  for (int j = 0; j < 32; ++j) { v[j] = bf2f(p[lane + j * 32]) * 0.125f; mx = fmaxf(mx, v[j]); }
  #pragma unroll
  for (int o = 16; o > 0; o >>= 1) mx = fmaxf(mx, __shfl_xor(mx, o, 32));
  float sum = 0.f;
  #pragma unroll
  for (int j = 0; j < 32; ++j) { v[j] = __expf(v[j] - mx); sum += v[j]; }
  #pragma unroll
  for (int o = 16; o > 0; o >>= 1) sum += __shfl_xor(sum, o, 32);
  const float inv = 1.f / sum;
  #pragma unroll
  for (int j = 0; j < 32; ++j) p[lane + j * 32] = f2bf(v[j] * inv);
}

// ---------------------------------------------------------------------------
// Bilinear resize (half-pixel centers) from (B,32,32,C) f32 -> (B,C,ts,ts)
// ---------------------------------------------------------------------------
__global__ void k_resize(const float* __restrict__ f, float* __restrict__ out,
                         int C, int ts, int total)
{
  int i = blockIdx.x * 256 + threadIdx.x;
  if (i >= total) return;
  int xo = i % ts;
  int yo = (i / ts) % ts;
  int c  = (i / (ts * ts)) % C;
  int b  = i / (ts * ts * C);
  const float scale = 32.f / (float)ts;
  float sy = ((float)yo + 0.5f) * scale - 0.5f;
  float sx = ((float)xo + 0.5f) * scale - 0.5f;
  sy = fminf(fmaxf(sy, 0.f), 31.f);
  sx = fminf(fmaxf(sx, 0.f), 31.f);
  int y0 = (int)sy, x0 = (int)sx;
  int y1 = y0 + 1 > 31 ? 31 : y0 + 1;
  int x1 = x0 + 1 > 31 ? 31 : x0 + 1;
  float wy = sy - (float)y0, wx = sx - (float)x0;
  const float* fb = f + (long)b * 1024 * C;
  float v00 = fb[(y0 * 32 + x0) * C + c], v01 = fb[(y0 * 32 + x1) * C + c];
  float v10 = fb[(y1 * 32 + x0) * C + c], v11 = fb[(y1 * 32 + x1) * C + c];
  out[i] = (1.f - wy) * ((1.f - wx) * v00 + wx * v01) + wy * ((1.f - wx) * v10 + wx * v11);
}

// ---------------------------------------------------------------------------
extern "C" void kernel_launch(void* const* d_in, const int* in_sizes, int n_in,
                              void* d_out, int out_size, void* d_ws, size_t ws_size,
                              hipStream_t stream)
{
  (void)in_sizes; (void)n_in; (void)out_size; (void)ws_size;
  const float* x       = (const float*)d_in[0];
  const float* patch_w = (const float*)d_in[1];
  const float* patch_b = (const float*)d_in[2];
  const float* pos     = (const float*)d_in[3];
  const float* ln1_w   = (const float*)d_in[4];
  const float* ln1_b   = (const float*)d_in[5];
  const float* qkv_w   = (const float*)d_in[6];
  const float* qkv_b   = (const float*)d_in[7];
  const float* proj_w  = (const float*)d_in[8];
  const float* proj_b  = (const float*)d_in[9];
  const float* ln2_w   = (const float*)d_in[10];
  const float* ln2_b   = (const float*)d_in[11];
  const float* fc1_w   = (const float*)d_in[12];
  const float* fc1_b   = (const float*)d_in[13];
  const float* fc2_w   = (const float*)d_in[14];
  const float* fc2_b   = (const float*)d_in[15];
  const float* norm_w  = (const float*)d_in[16];
  const float* norm_b  = (const float*)d_in[17];
  const float* fpn_w[4] = {(const float*)d_in[18], (const float*)d_in[20],
                           (const float*)d_in[22], (const float*)d_in[24]};
  const float* fpn_b[4] = {(const float*)d_in[19], (const float*)d_in[21],
                           (const float*)d_in[23], (const float*)d_in[25]};
  float* outp = (float*)d_out;

  // workspace carve-up (256B aligned)
  char* ws = (char*)d_ws;
  size_t off = 0;
  auto alloc = [&](size_t bytes) -> void* {
    void* p = (void*)(ws + off);
    off += (bytes + 255) & ~(size_t)255;
    return p;
  };
  bf16_t* actbf = (bf16_t*)alloc(2048UL * 768 * 2);        // LN out / im2row (reused)
  float*  t     = (float*) alloc(2048UL * 768 * 4);        // residual stream (f32)
  bf16_t* qkvb  = (bf16_t*)alloc(2048UL * 2304 * 2);       // qkv (bf16)
  bf16_t* S     = (bf16_t*)alloc(24UL * 1024 * 1024 * 2);  // attention scores/probs
  bf16_t* ao    = (bf16_t*)alloc(2048UL * 768 * 2);        // attention output
  bf16_t* g     = (bf16_t*)alloc(2048UL * 3072 * 2);       // MLP hidden (post-GELU)
  float*  fpn   = (float*) alloc(2048UL * 768 * 4);        // FPN features (bhwc)

  const dim3 blk(256);
  const long MB = (long)1024 * 1024;

  // patchify + patch-embed GEMM (+pos_embed) -> t (f32)
  {
    int tot = 2 * 1024 * 768;
    k_im2row<<<dim3((tot + 255) / 256), blk, 0, stream>>>(x, actbf, tot);
    k_gemm<float, false, false, false><<<dim3(768 / BN, 2048 / BM, 1), blk, 0, stream>>>(
        actbf, 768, 0, 0, patch_w, 768, 0, 0, patch_b, pos, nullptr,
        (void*)t, 768, 0, 0, 2048, 768, 768, 1);
  }

  const int out_ch[4]  = {96, 192, 384, 768};
  const long out_off[4] = {0, 3145728, 4718592, 5505024};

  for (int i = 0; i < 12; ++i) {
    // --- attention block ---
    k_ln<<<dim3(2048), blk, 0, stream>>>(t, ln1_w + i * 768, ln1_b + i * 768, actbf);
    k_gemm<float, false, true, false><<<dim3(2304 / BN, 16, 1), blk, 0, stream>>>(
        actbf, 768, 0, 0, qkv_w + (long)i * 768 * 2304, 2304, 0, 0,
        qkv_b + (long)i * 2304, nullptr, nullptr,
        (void*)qkvb, 2304, 0, 0, 2048, 2304, 768, 1);
    // S = Q @ K^T (batched over B*H = 24), bf16 logits
    k_gemm<bf16_t, true, true, false><<<dim3(1024 / BN, 1024 / BM, 24), blk, 0, stream>>>(
        qkvb, 2304, 1024L * 2304, 64,          // A = Q
        qkvb + 768, 2304, 1024L * 2304, 64,    // W = K (transposed access)
        nullptr, nullptr, nullptr,
        (void*)S, 1024, 12L * MB, MB, 1024, 1024, 64, 12);
    k_softmax<<<dim3(24576 / 8), blk, 0, stream>>>(S, 24576);
    // O = P @ V  -> ao (bf16, heads interleaved back to (B,N,768))
    k_gemm<bf16_t, false, true, false><<<dim3(1, 1024 / BM, 24), blk, 0, stream>>>(
        S, 1024, 12L * MB, MB,
        qkvb + 1536, 2304, 1024L * 2304, 64,   // W = V
        nullptr, nullptr, nullptr,
        (void*)ao, 768, 1024L * 768, 64, 1024, 64, 1024, 12);
    // t += O @ proj_w + proj_b
    k_gemm<float, false, false, false><<<dim3(768 / BN, 16, 1), blk, 0, stream>>>(
        ao, 768, 0, 0, proj_w + (long)i * 768 * 768, 768, 0, 0,
        proj_b + (long)i * 768, nullptr, t,
        (void*)t, 768, 0, 0, 2048, 768, 768, 1);

    // --- MLP block ---
    k_ln<<<dim3(2048), blk, 0, stream>>>(t, ln2_w + i * 768, ln2_b + i * 768, actbf);
    k_gemm<float, false, true, true><<<dim3(3072 / BN, 16, 1), blk, 0, stream>>>(
        actbf, 768, 0, 0, fc1_w + (long)i * 768 * 3072, 3072, 0, 0,
        fc1_b + (long)i * 3072, nullptr, nullptr,
        (void*)g, 3072, 0, 0, 2048, 3072, 768, 1);
    k_gemm<float, false, false, false><<<dim3(768 / BN, 16, 1), blk, 0, stream>>>(
        g, 3072, 0, 0, fc2_w + (long)i * 3072 * 768, 768, 0, 0,
        fc2_b + (long)i * 768, nullptr, t,
        (void*)t, 768, 0, 0, 2048, 768, 3072, 1);

    // --- FPN extraction at layers 3, 6, 9, 11 ---
    int e = (i == 3) ? 0 : (i == 6) ? 1 : (i == 9) ? 2 : (i == 11) ? 3 : -1;
    if (e >= 0) {
      k_ln<<<dim3(2048), blk, 0, stream>>>(t, norm_w, norm_b, actbf);
      const int oc = out_ch[e];
      k_gemm<float, false, false, false><<<dim3((oc + BN - 1) / BN, 16, 1), blk, 0, stream>>>(
          actbf, 768, 0, 0, fpn_w[e], oc, 0, 0, fpn_b[e], nullptr, nullptr,
          (void*)fpn, oc, 0, 0, 2048, oc, 768, 1);
      const int ts = 512 / (4 << e);
      const int tot = 2 * oc * ts * ts;
      k_resize<<<dim3((tot + 255) / 256), blk, 0, stream>>>(fpn, outp + out_off[e], oc, ts, tot);
    }
  }
}
